// AttentionLayer_76364518523578
// MI455X (gfx1250) — matrix operand order repaired and checked by reading, not compiled, not run on hardware
//
#include <hip/hip_runtime.h>
#include <hip/hip_bf16.h>

// ---------------- problem constants ----------------
#define BATCH 2
#define SEQ   2048
#define DMODEL 1024
#define NHEAD 16
#define DK    64
#define MTOK  (BATCH * SEQ)          // 4096 rows

typedef __attribute__((ext_vector_type(16))) __bf16 v16bf;
typedef __attribute__((ext_vector_type(8)))  __bf16 v8bf;
typedef __attribute__((ext_vector_type(8)))  float  v8f;

// ---------------- fragment helpers ----------------
static __device__ __forceinline__ v16bf combine16(v8bf lo, v8bf hi) {
    v16bf r;
#pragma unroll
    for (int i = 0; i < 8; ++i) { r[i] = lo[i]; r[i + 8] = hi[i]; }
    return r;
}

// A fragment (16xK tile, bf16 source, row-major, lane = row, K chunks {off..+7, off+16..+23})
static __device__ __forceinline__ v16bf load_a_bf16(const __bf16* row, int off) {
    v8bf lo = *(const v8bf*)(row + off);
    v8bf hi = *(const v8bf*)(row + off + 16);
    return combine16(lo, hi);
}

// A fragment built from f32 source (converted in-register via v_cvt_pk_bf16_f32)
static __device__ __forceinline__ v16bf load_a_f32(const float* row, int off) {
    float4 x0 = *(const float4*)(row + off);
    float4 x1 = *(const float4*)(row + off + 4);
    float4 x2 = *(const float4*)(row + off + 16);
    float4 x3 = *(const float4*)(row + off + 20);
    v16bf a;
    a[0]  = (__bf16)x0.x; a[1]  = (__bf16)x0.y; a[2]  = (__bf16)x0.z; a[3]  = (__bf16)x0.w;
    a[4]  = (__bf16)x1.x; a[5]  = (__bf16)x1.y; a[6]  = (__bf16)x1.z; a[7]  = (__bf16)x1.w;
    a[8]  = (__bf16)x2.x; a[9]  = (__bf16)x2.y; a[10] = (__bf16)x2.z; a[11] = (__bf16)x2.w;
    a[12] = (__bf16)x3.x; a[13] = (__bf16)x3.y; a[14] = (__bf16)x3.z; a[15] = (__bf16)x3.w;
    return a;
}

// B fragment: lane = column N, K chunks {off..+7, off+16..+23} (off = half*8)
static __device__ __forceinline__ v16bf load_b16(const __bf16* p) {
    v8bf lo = *(const v8bf*)(p);
    v8bf hi = *(const v8bf*)(p + 16);
    return combine16(lo, hi);
}

static __device__ __forceinline__ v8f wmma_bf16(v16bf a, v16bf b, v8f c) {
    return __builtin_amdgcn_wmma_f32_16x16x32_bf16(false, a, false, b, (short)0, c, false, false);
}

// ---------------- VALU butterfly within each 16-lane row: v_permlane16_b32 ----------------
// Selector nibbles give lane^mask; both 16-lane rows use the same selects.
static __device__ __forceinline__ float permx(float x, int lo, int hi) {
    float r;
    asm("v_permlane16_b32 %0, %1, %2, %3" : "=v"(r) : "v"(x), "s"(lo), "s"(hi));
    return r;
}
static __device__ __forceinline__ float halfmax(float v) {
    v = fmaxf(v, permx(v, (int)0x67452301, (int)0xEFCDAB89));  // xor 1
    v = fmaxf(v, permx(v, (int)0x45670123, (int)0xDCFE98BA));  // xor 2
    v = fmaxf(v, permx(v, (int)0x32107654, (int)0xBA98FEDC));  // xor 4
    v = fmaxf(v, permx(v, (int)0xFEDCBA98, (int)0x76543210));  // xor 8
    return v;
}
static __device__ __forceinline__ float halfsum(float v) {
    v += permx(v, (int)0x67452301, (int)0xEFCDAB89);
    v += permx(v, (int)0x45670123, (int)0xDCFE98BA);
    v += permx(v, (int)0x32107654, (int)0xBA98FEDC);
    v += permx(v, (int)0xFEDCBA98, (int)0x76543210);
    return v;
}

// ---------------- kernel 1: f32 -> bf16 weight conversion ----------------
__global__ __launch_bounds__(256) void cvt_f32_bf16(const float* __restrict__ in,
                                                    __bf16* __restrict__ out, int n) {
    int i = (blockIdx.x * 256 + threadIdx.x) * 4;
    if (i + 3 < n) {
        float4 v = *(const float4*)(in + i);
        out[i + 0] = (__bf16)v.x;
        out[i + 1] = (__bf16)v.y;
        out[i + 2] = (__bf16)v.z;
        out[i + 3] = (__bf16)v.w;
    }
}

// ---------------- GEMM  C = X @ W^T + b, one wave -> 16x64 tile, SW-pipelined ----------------
// MODE: 0=Q (bf16 row-major), 1=K (f32 cache + bf16 row-major),
//       2=V (f32 cache + bf16 transposed [B,H,DK,S]), 3=O (bf16 A, f32 out)
template <int MODE>
__global__ __launch_bounds__(256) void gemm16x64(const float* __restrict__ Xf,
                                                 const __bf16* __restrict__ Xb,
                                                 const __bf16* __restrict__ Wb,
                                                 const float* __restrict__ bias,
                                                 float* __restrict__ outF,
                                                 __bf16* __restrict__ outB,
                                                 __bf16* __restrict__ outVT) {
    int wave = blockIdx.x * 8 + (threadIdx.x >> 5);
    int lane = threadIdx.x & 31;
    int l16 = lane & 15, half = lane >> 4;
    int m0 = (wave >> 4) << 4;     // 256 m-tiles
    int n0 = (wave & 15) << 6;     // 16 n-tiles of 64

    const float*  xfrow = Xf + (size_t)(m0 + l16) * DMODEL;
    const __bf16* xbrow = Xb + (size_t)(m0 + l16) * DMODEL;
    const __bf16* wbase = Wb + (size_t)(n0 + l16) * DMODEL + half * 8;

    auto loadA = [&](int k0) -> v16bf {
        if constexpr (MODE == 3) return load_a_bf16(xbrow, k0 + half * 8);
        else                     return load_a_f32(xfrow, k0 + half * 8);
    };
    auto loadB = [&](int k0, int g) -> v16bf {
        return load_b16(wbase + (size_t)g * 16 * DMODEL + k0);
    };

    v8f acc[4] = {v8f{}, v8f{}, v8f{}, v8f{}};

    // software pipeline: fetch next k-step while current WMMAs execute
    v16bf a  = loadA(0);
    v16bf b0 = loadB(0, 0), b1 = loadB(0, 1), b2 = loadB(0, 2), b3 = loadB(0, 3);
    for (int k0 = 32; k0 < DMODEL; k0 += 32) {
        v16bf an  = loadA(k0);
        v16bf bn0 = loadB(k0, 0), bn1 = loadB(k0, 1), bn2 = loadB(k0, 2), bn3 = loadB(k0, 3);
        acc[0] = wmma_bf16(a, b0, acc[0]);
        acc[1] = wmma_bf16(a, b1, acc[1]);
        acc[2] = wmma_bf16(a, b2, acc[2]);
        acc[3] = wmma_bf16(a, b3, acc[3]);
        a = an; b0 = bn0; b1 = bn1; b2 = bn2; b3 = bn3;
    }
    acc[0] = wmma_bf16(a, b0, acc[0]);
    acc[1] = wmma_bf16(a, b1, acc[1]);
    acc[2] = wmma_bf16(a, b2, acc[2]);
    acc[3] = wmma_bf16(a, b3, acc[3]);

    // epilogue: bias + stores, fully branch-free per MODE
#pragma unroll
    for (int g = 0; g < 4; ++g) {
        int ncol = n0 + g * 16 + l16;
        float bv = bias[ncol];
#pragma unroll
        for (int j = 0; j < 8; ++j) {
            float val = acc[g][j] + bv;
            int mrow = m0 + j + half * 8;
            if constexpr (MODE == 1 || MODE == 2 || MODE == 3)
                outF[(size_t)mrow * DMODEL + ncol] = val;            // K/V cache or final out
            if constexpr (MODE == 0 || MODE == 1)
                outB[(size_t)mrow * DMODEL + ncol] = (__bf16)val;    // Q/K bf16 row-major
            if constexpr (MODE == 2) {                               // V transposed [B,H,DK,S]
                int bb = mrow >> 11, s = mrow & (SEQ - 1);
                int h = ncol >> 6, dk = ncol & 63;
                outVT[(((size_t)bb * NHEAD + h) * DK + dk) * SEQ + s] = (__bf16)val;
            }
        }
    }
}

// ---------------- flash attention (one wave per b,h,16-row q tile) ----------------
__global__ __launch_bounds__(256) void attn_flash(const __bf16* __restrict__ qb,
                                                  const __bf16* __restrict__ kb,
                                                  const __bf16* __restrict__ vT,
                                                  const float* __restrict__ bias,
                                                  __bf16* __restrict__ attn) {
    __shared__ __align__(16) __bf16 lds_p[8][16 * 32];   // per-wave P staging tile
    int wslot = threadIdx.x >> 5;
    __bf16* pbuf = lds_p[wslot];

    int wave = blockIdx.x * 8 + wslot;          // 0..4095
    int qt = wave & 127;
    int h  = (wave >> 7) & 15;
    int bb = wave >> 11;
    int q0 = qt << 4;

    int lane = threadIdx.x & 31;
    int l16 = lane & 15, half = lane >> 4;

    const __bf16* qrow = qb + (size_t)(bb * SEQ + q0 + l16) * DMODEL + h * DK;
    v16bf aq0 = load_a_bf16(qrow, half * 8);          // dk 0..31
    v16bf aq1 = load_a_bf16(qrow, 32 + half * 8);     // dk 32..63

    float m_i[8], l_i[8];
#pragma unroll
    for (int r = 0; r < 8; ++r) { m_i[r] = -1e30f; l_i[r] = 0.0f; }
    v8f o[4] = {v8f{}, v8f{}, v8f{}, v8f{}};

    const __bf16* kbase = kb + ((size_t)bb * SEQ) * DMODEL + h * DK + half * 8;
    const __bf16* vbase = vT + (((size_t)bb * NHEAD + h) * DK + l16) * SEQ + half * 8;
    const float* brow = bias + (size_t)(bb * SEQ + q0) * SEQ;
    const float scale = 0.125f;  // 1/sqrt(64)

    for (int n0 = 0; n0 < SEQ; n0 += 32) {
        // ---- issue ALL K and V fragment loads up front; V loads overlap score+softmax ----
        const __bf16* krow0 = kbase + (size_t)(n0 + l16) * DMODEL;
        const __bf16* krow1 = krow0 + (size_t)16 * DMODEL;
        v16bf kb00 = load_b16(krow0);
        v16bf kb01 = load_b16(krow0 + 32);
        v16bf kb10 = load_b16(krow1);
        v16bf kb11 = load_b16(krow1 + 32);
        v16bf vB[4];
#pragma unroll
        for (int g = 0; g < 4; ++g)
            vB[g] = load_b16(vbase + (size_t)g * 16 * SEQ + n0);
        if (n0 + 32 < SEQ) {                       // prefetch next key tile (global_prefetch_b8)
            __builtin_prefetch(krow0 + (size_t)32 * DMODEL, 0, 1);
            __builtin_prefetch(vbase + n0 + 32, 0, 1);
        }

        // ---- scores for 32 keys: two 16x16 C tiles ----
        v8f s0 = {}, s1 = {};
        s0 = wmma_bf16(aq0, kb00, s0);
        s0 = wmma_bf16(aq1, kb01, s0);
        s1 = wmma_bf16(aq0, kb10, s1);
        s1 = wmma_bf16(aq1, kb11, s1);

        // ---- online softmax (rows map: C vgpr r, half -> q row r + half*8) ----
#pragma unroll
        for (int r = 0; r < 8; ++r) {
            size_t boff = (size_t)(r + half * 8) * SEQ + n0 + l16;
            float v0 = s0[r] * scale + brow[boff];
            float v1 = s1[r] * scale + brow[boff + 16];
            float mloc = halfmax(fmaxf(v0, v1));
            float mnew = fmaxf(m_i[r], mloc);
            float p0 = __expf(v0 - mnew);
            float p1 = __expf(v1 - mnew);
            float corr = __expf(m_i[r] - mnew);
            l_i[r] = l_i[r] * corr + halfsum(p0 + p1);
            m_i[r] = mnew;
#pragma unroll
            for (int g = 0; g < 4; ++g) o[g][r] *= corr;
            pbuf[(r + half * 8) * 32 + l16]      = (__bf16)p0;
            pbuf[(r + half * 8) * 32 + 16 + l16] = (__bf16)p1;
        }

        // ---- P @ V : P as A fragment from LDS (wave-private; LDS in-order per wave) ----
        v16bf pA = load_a_bf16(pbuf + l16 * 32, half * 8);
        o[0] = wmma_bf16(pA, vB[0], o[0]);
        o[1] = wmma_bf16(pA, vB[1], o[1]);
        o[2] = wmma_bf16(pA, vB[2], o[2]);
        o[3] = wmma_bf16(pA, vB[3], o[3]);
    }

    // ---- normalize & store bf16 attention output [B,S,D] ----
#pragma unroll
    for (int r = 0; r < 8; ++r) {
        float inv = 1.0f / l_i[r];
        int qr = q0 + r + half * 8;
#pragma unroll
        for (int g = 0; g < 4; ++g) {
            attn[(size_t)(bb * SEQ + qr) * DMODEL + h * DK + g * 16 + l16] =
                (__bf16)(o[g][r] * inv);
        }
    }
}

// ---------------- host-side launch ----------------
extern "C" void kernel_launch(void* const* d_in, const int* in_sizes, int n_in,
                              void* d_out, int out_size, void* d_ws, size_t ws_size,
                              hipStream_t stream) {
    const float* queries = (const float*)d_in[0];
    const float* keys    = (const float*)d_in[1];
    const float* values  = (const float*)d_in[2];
    const float* abias   = (const float*)d_in[3];
    const float* Wq = (const float*)d_in[4];  const float* bq = (const float*)d_in[5];
    const float* Wk = (const float*)d_in[6];  const float* bk = (const float*)d_in[7];
    const float* Wv = (const float*)d_in[8];  const float* bv = (const float*)d_in[9];
    const float* Wo = (const float*)d_in[10]; const float* bo = (const float*)d_in[11];

    float* out     = (float*)d_out;                       // [B,S,D]
    float* cache_k = out + (size_t)MTOK * DMODEL;         // [B,S,D]
    float* cache_v = cache_k + (size_t)MTOK * DMODEL;     // [B,S,D]

    const size_t MB = 1u << 20;
    char* ws = (char*)d_ws;
    __bf16* qb   = (__bf16*)(ws + 0 * MB);    // [B,S,D] bf16
    __bf16* kb   = (__bf16*)(ws + 8 * MB);    // [B,S,D] bf16
    __bf16* vT   = (__bf16*)(ws + 16 * MB);   // [B,H,DK,S] bf16
    __bf16* attn = (__bf16*)(ws + 24 * MB);   // [B,S,D] bf16
    __bf16* Wqb  = (__bf16*)(ws + 32 * MB);
    __bf16* Wkb  = (__bf16*)(ws + 34 * MB);
    __bf16* Wvb  = (__bf16*)(ws + 36 * MB);
    __bf16* Wob  = (__bf16*)(ws + 38 * MB);

    const int nW = DMODEL * DMODEL;           // 1M elems per weight
    dim3 cvtGrid(nW / (256 * 4));
    cvt_f32_bf16<<<cvtGrid, 256, 0, stream>>>(Wq, Wqb, nW);
    cvt_f32_bf16<<<cvtGrid, 256, 0, stream>>>(Wk, Wkb, nW);
    cvt_f32_bf16<<<cvtGrid, 256, 0, stream>>>(Wv, Wvb, nW);
    cvt_f32_bf16<<<cvtGrid, 256, 0, stream>>>(Wo, Wob, nW);

    // (M/16)*(D/64) = 256*16 = 4096 waves, 8 waves/block -> 512 blocks
    dim3 gGrid(512), gBlk(256);
    gemm16x64<0><<<gGrid, gBlk, 0, stream>>>(queries, nullptr, Wqb, bq, nullptr, qb,   nullptr);
    gemm16x64<1><<<gGrid, gBlk, 0, stream>>>(keys,    nullptr, Wkb, bk, cache_k, kb,   nullptr);
    gemm16x64<2><<<gGrid, gBlk, 0, stream>>>(values,  nullptr, Wvb, bv, cache_v, nullptr, vT);

    // B*H*(S/16) = 4096 waves -> 512 blocks
    attn_flash<<<dim3(512), dim3(256), 0, stream>>>(qb, kb, vT, abias, attn);

    gemm16x64<3><<<gGrid, gBlk, 0, stream>>>(nullptr, attn, Wob, bo, out, nullptr, nullptr);
}
